// TimeSelfAttention_70033736728730
// MI455X (gfx1250) — matrix operand order repaired and checked by reading, not compiled, not run on hardware
//
#include <hip/hip_runtime.h>

// ---------------------------------------------------------------------------
// Sliding-window block-banded attention for MI455X (gfx1250, wave32, WMMA).
// B=8 S=2048 DM=512 H=8 DK=64 BLK=R=128, window = blocks [b-1,b,b+1] (384 keys).
// Matmuls on v_wmma_f32_16x16x32_f16; K window staged to LDS with
// global_load_async_to_lds_b128 (ASYNCcnt); V consumed pre-transposed.
// ---------------------------------------------------------------------------

typedef _Float16 h16;
typedef __attribute__((ext_vector_type(16))) _Float16 v16h;
typedef __attribute__((ext_vector_type(8)))  float    v8f;

#define BB    8
#define SS    2048
#define DM    512
#define HH    8
#define DK    64
#define BLKK  128
#define NB    16
#define WIN   384          // 3*BLKK
#define MTOT  (BB*SS)      // 16384

union FragU { uint4 q[2]; v16h h; };

__device__ __forceinline__ v8f wmma16(v16h a, v16h b, v8f c) {
  return __builtin_amdgcn_wmma_f32_16x16x32_f16(false, a, false, b, (short)0, c,
                                                false, false);
}

// A-fragment 16x32 f16: lane m = lane&15, K split {0-7,16-23}/{8-15,24-31}
// over half-waves; pairs along K contiguous -> two 16B loads per lane.
__device__ __forceinline__ v16h load_frag_a(const h16* p, int ld) {
  const int lane = threadIdx.x & 31;
  const h16* r = p + (size_t)(lane & 15) * ld + ((lane >> 4) << 3);
  FragU f;
  f.q[0] = *(const uint4*)(r);
  f.q[1] = *(const uint4*)(r + 16);
  return f.h;
}

// B-fragment 32x16 f16 read from an N-major [n][k] array: lane n = lane&15,
// K split {0-7,16-23}/{8-15,24-31} over half-waves; pairs along K contiguous.
__device__ __forceinline__ v16h load_frag_b(const h16* p, int ld) {
  const int lane = threadIdx.x & 31;
  const h16* r = p + (size_t)(lane & 15) * ld + ((lane >> 4) << 3);
  FragU f;
  f.q[0] = *(const uint4*)(r);
  f.q[1] = *(const uint4*)(r + 16);
  return f.h;
}

// Async global -> LDS 128-bit copy (CDNA5 VGLOBAL async op, ASYNCcnt).
__device__ __forceinline__ void async_copy_b128(unsigned lds_off,
                                                const void* gptr) {
  asm volatile("global_load_async_to_lds_b128 %0, %1, off"
               :: "v"(lds_off), "v"(gptr) : "memory");
}

// ---------------------------------------------------------------------------
// fp32 -> f16 convert (X)
// ---------------------------------------------------------------------------
__global__ void swa_cvt_f16(const float* __restrict__ in, h16* __restrict__ out,
                            int n) {
  int i = (blockIdx.x * blockDim.x + threadIdx.x) * 4;
  if (i + 3 < n) {
    float4 v = *(const float4*)(in + i);
    out[i + 0] = (h16)v.x; out[i + 1] = (h16)v.y;
    out[i + 2] = (h16)v.z; out[i + 3] = (h16)v.w;
  }
}

// Wt[n][k] = (f16) W[k][n]  (K=N=512) so GEMM B-fragments are k-contiguous.
__global__ void swa_transpose_w(const float* __restrict__ W, h16* __restrict__ Wt) {
  int idx = blockIdx.x * blockDim.x + threadIdx.x;   // idx = n*512 + k
  int n = idx >> 9, k = idx & 511;
  Wt[idx] = (h16)W[(size_t)k * DM + n];
}

// Vt[b,h,d,s] = Vh[b,s,h,d]  so P.V B-fragments are key-contiguous.
__global__ void swa_vt(const h16* __restrict__ Vh, h16* __restrict__ Vt) {
  int idx = blockIdx.x * blockDim.x + threadIdx.x;   // B*H*DK*S = 8388608
  int s = idx & (SS - 1);
  int d = (idx >> 11) & (DK - 1);
  int h = (idx >> 17) & (HH - 1);
  int b = idx >> 20;
  Vt[idx] = Vh[((size_t)b * SS + s) * DM + h * DK + d];
}

// ---------------------------------------------------------------------------
// GEMM: C[M,N] = A[M,K](f16) x Bt[N,K](f16), fp32 accumulate.
// 128x128 workgroup tile, 8 waves, each wave 32(M)x64(N) = 2x4 WMMA accums.
// ---------------------------------------------------------------------------
template <bool F16OUT>
__global__ __launch_bounds__(256) void swa_gemm(const h16* __restrict__ A,
                                                const h16* __restrict__ Bt,
                                                h16* __restrict__ Ch,
                                                float* __restrict__ Cf,
                                                int M, int N, int K) {
  const int wave = threadIdx.x >> 5;
  const int lane = threadIdx.x & 31;
  const int m0 = blockIdx.y * 128 + (wave & 3) * 32;
  const int n0 = blockIdx.x * 128 + (wave >> 2) * 64;

  v8f acc[2][4] = {{{}, {}, {}, {}}, {{}, {}, {}, {}}};
  for (int kc = 0; kc < K; kc += 32) {
    if (kc + 32 < K) {   // keep the WMMA pipe fed (emits global_prefetch_b8)
      __builtin_prefetch(A + (size_t)m0 * K + kc + 32, 0, 1);
      __builtin_prefetch(Bt + (size_t)n0 * K + kc + 32, 0, 1);
    }
    v16h a0 = load_frag_a(A + (size_t)m0 * K + kc, K);
    v16h a1 = load_frag_a(A + (size_t)(m0 + 16) * K + kc, K);
#pragma unroll
    for (int nt = 0; nt < 4; ++nt) {
      v16h b = load_frag_b(Bt + (size_t)(n0 + nt * 16) * K + kc, K);
      acc[0][nt] = wmma16(a0, b, acc[0][nt]);
      acc[1][nt] = wmma16(a1, b, acc[1][nt]);
    }
  }
  // C layout: VGPR r -> row r (lanes 0-15) / r+8 (lanes 16-31), col = lane&15.
  const int rowh = (lane >> 4) << 3;
  const int col  = lane & 15;
#pragma unroll
  for (int mt = 0; mt < 2; ++mt)
#pragma unroll
    for (int nt = 0; nt < 4; ++nt)
#pragma unroll
      for (int r = 0; r < 8; ++r) {
        size_t o = (size_t)(m0 + mt * 16 + rowh + r) * N + (n0 + nt * 16 + col);
        if constexpr (F16OUT) Ch[o] = (h16)acc[mt][nt][r];
        else                  Cf[o] = acc[mt][nt][r];
      }
}

// ---------------------------------------------------------------------------
// RoPE on Q,K in place (f16 storage, fp32 math). Layout [B,S,H,DK].
// ---------------------------------------------------------------------------
__global__ void swa_rope(h16* __restrict__ Q, h16* __restrict__ K, int npairs) {
  int p = blockIdx.x * blockDim.x + threadIdx.x;
  if (p >= npairs) return;
  int i    = p & 31;           // freq index within head (half = 32)
  int head = (p >> 5) & 7;
  int row  = p >> 8;           // b*S + s
  int s    = row & (SS - 1);
  float inv = __expf(-(float)i * (9.210340372f / 32.0f));  // theta^(-i/32)
  float ang = (float)s * inv;
  float sn, cs;
  __sincosf(ang, &sn, &cs);
  size_t e0 = (size_t)row * DM + head * DK + i;
  size_t e1 = e0 + 32;
  float q0 = (float)Q[e0], q1 = (float)Q[e1];
  Q[e0] = (h16)(q0 * cs - q1 * sn);
  Q[e1] = (h16)(q1 * cs + q0 * sn);
  float k0 = (float)K[e0], k1 = (float)K[e1];
  K[e0] = (h16)(k0 * cs - k1 * sn);
  K[e1] = (h16)(k1 * cs + k0 * sn);
}

// ---------------------------------------------------------------------------
// Attention: one workgroup per (b, h, qblock). 8 waves x 16 query rows.
// K window async-staged to LDS once (no transpose needed: contraction dim d is
// contiguous). V fragments straight from pre-transposed global Vt. P bounced
// through per-wave LDS scratch to become A-fragments. Masked scores are -3e4,
// so exp() underflows to exactly 0 and clamped/garbage K,V rows contribute 0.
// ---------------------------------------------------------------------------
__global__ __launch_bounds__(256) void swa_attn(const h16* __restrict__ Q,
                                                const h16* __restrict__ K,
                                                const h16* __restrict__ Vt,
                                                h16* __restrict__ O) {
  __shared__ h16 lds_k[WIN * DK];        // 48 KiB  K[key][d]
  __shared__ h16 lds_p[8 * 16 * 32];     //  8 KiB  per-wave P chunk scratch

  const int blk = blockIdx.x, h = blockIdx.y, b = blockIdx.z;
  const int tid = threadIdx.x, wave = tid >> 5, lane = tid & 31;
  const size_t base = ((size_t)b * SS * HH + h) * DK;       // Q/K/O: + s*512 + d
  const h16* vtb = Vt + ((size_t)b * HH + h) * DK * SS;     // Vt: + d*S + s
  const int kbase = (blk - 1) * BLKK;

  // ---- async-stage K window into LDS (rows clamped; mask fixes values) ----
  for (int idx = tid; idx < WIN * 8; idx += 256) {
    int key = idx >> 3;
    int c   = (idx & 7) << 3;                       // halves within row
    int kgc = min(max(kbase + key, 0), SS - 1);
    async_copy_b128((unsigned)(uintptr_t)&lds_k[key * DK + c],
                    K + base + (size_t)kgc * DM + c);
  }
  asm volatile("s_wait_asynccnt 0x0" ::: "memory");
  __syncthreads();

  // ---- scores: S[16q x 384k] held in 24 register accumulators ----
  const int qrow0 = blk * BLKK + wave * 16;
  v16h aq0 = load_frag_a(Q + base + (size_t)qrow0 * DM + 0, DM);
  v16h aq1 = load_frag_a(Q + base + (size_t)qrow0 * DM + 32, DM);

  const int col  = lane & 15;
  const int rowh = (lane >> 4) << 3;
  const int kgl  = kbase + col;          // this lane's key column base

  v8f sc[24];
#pragma unroll
  for (int kt = 0; kt < 24; ++kt) {
    v16h b0 = load_frag_b(lds_k + (kt * 16) * DK + 0, DK);
    v16h b1 = load_frag_b(lds_k + (kt * 16) * DK + 32, DK);
    v8f c = {};
    c = wmma16(aq0, b0, c);
    c = wmma16(aq1, b1, c);
    int kg = kgl + kt * 16;
#pragma unroll
    for (int rr = 0; rr < 8; ++rr) {
      int qg = qrow0 + rowh + rr;
      int d  = kg - qg;
      bool valid = (d <= 128) && (d >= -128) && (kg >= 0) && (kg < SS);
      c[rr] = valid ? c[rr] : -3.0e4f;
    }
    sc[kt] = c;
  }

  // ---- row-wise softmax stats (reduce across 16-lane halves) ----
  float rsum[8];
#pragma unroll
  for (int r = 0; r < 8; ++r) {
    float m = -3.0e4f;
#pragma unroll
    for (int kt = 0; kt < 24; ++kt) m = fmaxf(m, sc[kt][r]);
#pragma unroll
    for (int off = 8; off; off >>= 1) m = fmaxf(m, __shfl_xor(m, off, 32));
    float s = 0.0f;
#pragma unroll
    for (int kt = 0; kt < 24; ++kt) {
      float e = __expf(sc[kt][r] - m);
      sc[kt][r] = e;
      s += e;
    }
#pragma unroll
    for (int off = 8; off; off >>= 1) s += __shfl_xor(s, off, 32);
    rsum[r] = s;
  }

  // ---- O = P.V : bounce P chunks through per-wave LDS scratch ----
  h16* pw = lds_p + wave * (16 * 32);
  v8f oc[4] = {{}, {}, {}, {}};
#pragma unroll
  for (int kc = 0; kc < 12; ++kc) {
#pragma unroll
    for (int half = 0; half < 2; ++half) {
      int kt = 2 * kc + half;
#pragma unroll
      for (int r = 0; r < 8; ++r)
        pw[(rowh + r) * 32 + half * 16 + col] = (h16)sc[kt][r];
    }
    asm volatile("s_wait_dscnt 0x0" ::: "memory");  // LDS RAW, same wave
    v16h ap = load_frag_a(pw, 32);
#pragma unroll
    for (int nt = 0; nt < 4; ++nt) {
      // Vt row d = nt*16 + lane&15; keys kc*32... (may dip before/after the
      // (b,h) slab: stays inside d_ws, and P there is exactly 0).
      v16h bv = load_frag_b(vtb + (ptrdiff_t)(nt * 16) * SS + kbase + kc * 32,
                            SS);
      oc[nt] = wmma16(ap, bv, oc[nt]);
    }
  }

  // ---- normalize + store (rsum layout matches C layout rows) ----
#pragma unroll
  for (int nt = 0; nt < 4; ++nt)
#pragma unroll
    for (int r = 0; r < 8; ++r) {
      int srow = qrow0 + rowh + r;
      float v = oc[nt][r] / rsum[r];
      O[base + (size_t)srow * DM + nt * 16 + col] = (h16)v;
    }
}

// ---------------------------------------------------------------------------
extern "C" void kernel_launch(void* const* d_in, const int* in_sizes, int n_in,
                              void* d_out, int out_size, void* d_ws, size_t ws_size,
                              hipStream_t stream) {
  const float* X  = (const float*)d_in[0];
  const float* Wq = (const float*)d_in[1];
  const float* Wk = (const float*)d_in[2];
  const float* Wv = (const float*)d_in[3];
  const float* Wo = (const float*)d_in[4];
  float* out = (float*)d_out;

  const int M = MTOT, N = DM, K = DM;

  char* ws = (char*)d_ws;
  h16* Xh  = (h16*)ws; ws += (size_t)M * K * 2;   // 16 MiB
  h16* Wqt = (h16*)ws; ws += (size_t)N * K * 2;
  h16* Wkt = (h16*)ws; ws += (size_t)N * K * 2;
  h16* Wvt = (h16*)ws; ws += (size_t)N * K * 2;
  h16* Wot = (h16*)ws; ws += (size_t)N * K * 2;
  h16* Qh  = (h16*)ws; ws += (size_t)M * N * 2;   // 16 MiB each
  h16* Kh  = (h16*)ws; ws += (size_t)M * N * 2;
  h16* Vh  = (h16*)ws; ws += (size_t)M * N * 2;
  h16* Vtr = (h16*)ws; ws += (size_t)M * N * 2;   // keep AOh after: OOB slack
  h16* AOh = (h16*)ws; ws += (size_t)M * N * 2;

  swa_cvt_f16<<<(M * K) / 1024, 256, 0, stream>>>(X, Xh, M * K);
  swa_transpose_w<<<(N * K) / 256, 256, 0, stream>>>(Wq, Wqt);
  swa_transpose_w<<<(N * K) / 256, 256, 0, stream>>>(Wk, Wkt);
  swa_transpose_w<<<(N * K) / 256, 256, 0, stream>>>(Wv, Wvt);
  swa_transpose_w<<<(N * K) / 256, 256, 0, stream>>>(Wo, Wot);

  dim3 gg(N / 128, M / 128);
  swa_gemm<true><<<gg, 256, 0, stream>>>(Xh, Wqt, Qh, nullptr, M, N, K);
  swa_gemm<true><<<gg, 256, 0, stream>>>(Xh, Wkt, Kh, nullptr, M, N, K);
  swa_gemm<true><<<gg, 256, 0, stream>>>(Xh, Wvt, Vh, nullptr, M, N, K);

  const int npairs = M * HH * 32;   // 4,194,304
  swa_rope<<<npairs / 256, 256, 0, stream>>>(Qh, Kh, npairs);
  swa_vt<<<(M * N) / 256, 256, 0, stream>>>(Vh, Vtr);

  swa_attn<<<dim3(NB, HH, BB), 256, 0, stream>>>(Qh, Kh, Vtr, AOh);

  swa_gemm<false><<<gg, 256, 0, stream>>>(AOh, Wot, nullptr, out, M, N, K);
}
